// ResGNN_50629074485391
// MI455X (gfx1250) — matrix operand order repaired and checked by reading, not compiled, not run on hardware
//
#include <hip/hip_runtime.h>
#include <hip/hip_bf16.h>

// ---------------------------------------------------------------------------
// ResGNN for MI455X (gfx1250, wave32).
// GEMMs: v_wmma_f32_16x16x32_bf16, f32 accumulate.  Activations and weights
// are pre-converted to bf16 outside the GEMM inner loop; weights are packed
// in exact WMMA B-fragment order so each lane loads its fragment as one
// contiguous 32B chunk.  Each wave computes a 32x64 output block: 2 M-tiles
// x 4 N-tiles, so 4 B fragments feed 8 WMMAs per 32-wide K-chunk.
// Graph-conv scatter uses non-returning global_atomic_add_f32 (accumulator
// lives in the 192MB L2); edge-index stream prefetched (global_prefetch_b8).
// ---------------------------------------------------------------------------

typedef unsigned short ushort;
typedef __attribute__((ext_vector_type(8)))  unsigned short v8us;
typedef __attribute__((ext_vector_type(16))) unsigned short v16us;
typedef __attribute__((ext_vector_type(16))) __bf16         v16bf;
typedef __attribute__((ext_vector_type(8)))  float          v8f;

#define GN_N   100000
#define GN_E   3200000
#define GN_G   64
#define GN_L   4
#define GN_D   64
#define GN_H1  256
#define GN_H2  128
#define GN_EPS 1e-5f

__device__ __forceinline__ ushort f2bf(float f) {
  // round-to-nearest-even f32 -> bf16
  unsigned int u = __float_as_uint(f);
  u += 0x7FFFu + ((u >> 16) & 1u);
  return (ushort)(u >> 16);
}
__device__ __forceinline__ float bf2f(ushort u) {
  return __uint_as_float(((unsigned int)u) << 16);
}

// ---------------------------------------------------------------------------
// elementwise f32 -> bf16 conversion
// ---------------------------------------------------------------------------
__global__ void k_f2bf(const float* __restrict__ src, ushort* __restrict__ dst,
                       int n) {
  int i = blockIdx.x * blockDim.x + threadIdx.x;
  if (i < n) dst[i] = f2bf(src[i]);
}

// ---------------------------------------------------------------------------
// Pack a row-major f32 weight W[K x Nout] into bf16 WMMA B-fragment order:
//   P[((nt*(K/32) + kc)*32 + lane)*16 + e] = bf16(W[k][n])
//   k = kc*32 + (lane<16 ? 0 : 8) + e + (e<8 ? 0 : 8),  n = nt*16 + (lane&15)
// One thread per packed element.
// ---------------------------------------------------------------------------
__global__ void k_pack_w(const float* __restrict__ W, ushort* __restrict__ P,
                         int K, int Nout) {
  int i = blockIdx.x * blockDim.x + threadIdx.x;
  if (i >= K * Nout) return;
  int e       = i & 15;
  int lane    = (i >> 4) & 31;
  int kchunks = K >> 5;
  int kc      = (i >> 9) % kchunks;
  int nt      = (i >> 9) / kchunks;
  int k = (kc << 5) + ((lane & 16) ? 8 : 0) + e + ((e & 8) ? 8 : 0);
  int n = (nt << 4) + (lane & 15);
  P[i] = f2bf(W[(size_t)k * Nout + n]);
}

// ---------------------------------------------------------------------------
// GEMM: C[M x Nout] = act(A[M x K] * B[K x Nout] + bias)
//   A: bf16 row-major (lda elements), B: packed bf16 fragments (k_pack_w)
//   One wave per 32x64 output block (2 M-tiles x 4 N-tiles => 8 WMMAs share
//   4 B fragments per K-chunk).
//   Requires M%32==0, K%32==0, Nout%64==0.  grid = (M/32, Nout/64), block=32.
// ---------------------------------------------------------------------------
template <bool OUT_BF16>
__global__ void wmma_gemm_bf16(const ushort* __restrict__ A, int lda,
                               const ushort* __restrict__ Bp,
                               const float* __restrict__ bias,
                               void* __restrict__ Cout, int ldc,
                               int K, int relu) {
  const int lane    = threadIdx.x & 31;
  const int half    = lane >> 4;      // 0: lanes 0-15, 1: lanes 16-31
  const int l15     = lane & 15;
  const int kA      = half * 8;       // A k-base inside 32-chunk
  const int kchunks = K >> 5;
  const int nt0     = blockIdx.y * 4;

  v8f acc[2][4] = {};
  const ushort* arow0 = A + (size_t)(blockIdx.x * 32 + l15) * lda + kA;
  const ushort* arow1 = arow0 + (size_t)16 * lda;
  const ushort* bbase = Bp + ((size_t)nt0 * kchunks * 32 + lane) * 16;
  const size_t  bTile = (size_t)kchunks * 32 * 16;   // elements per n-tile

  for (int kc = 0; kc < kchunks; ++kc) {
    // A fragments: two contiguous 16B runs per m-tile (k = kA+0..7, kA+16..23)
    v8us a00 = *(const v8us*)(arow0);
    v8us a01 = *(const v8us*)(arow0 + 16);
    v8us a10 = *(const v8us*)(arow1);
    v8us a11 = *(const v8us*)(arow1 + 16);
    arow0 += 32;
    arow1 += 32;
    v16bf a0 = __builtin_bit_cast(v16bf,
        __builtin_shufflevector(a00, a01, 0, 1, 2, 3, 4, 5, 6, 7,
                                8, 9, 10, 11, 12, 13, 14, 15));
    v16bf a1 = __builtin_bit_cast(v16bf,
        __builtin_shufflevector(a10, a11, 0, 1, 2, 3, 4, 5, 6, 7,
                                8, 9, 10, 11, 12, 13, 14, 15));

    const ushort* bptr = bbase + (size_t)kc * (32 * 16);
    v16bf bfrag[4];
#pragma unroll
    for (int t = 0; t < 4; ++t)
      bfrag[t] = *(const v16bf*)(bptr + (size_t)t * bTile);

#pragma unroll
    for (int t = 0; t < 4; ++t) {
      acc[0][t] = __builtin_amdgcn_wmma_f32_16x16x32_bf16(
          false, a0, false, bfrag[t], (short)0, acc[0][t], false, false);
      acc[1][t] = __builtin_amdgcn_wmma_f32_16x16x32_bf16(
          false, a1, false, bfrag[t], (short)0, acc[1][t], false, false);
    }
  }

#pragma unroll
  for (int m = 0; m < 2; ++m) {
    const int mBase = blockIdx.x * 32 + m * 16 + half * 8;
#pragma unroll
    for (int t = 0; t < 4; ++t) {
      const int ncol = (nt0 + t) * 16 + l15;
      const float bv = bias ? bias[ncol] : 0.0f;
#pragma unroll
      for (int v = 0; v < 8; ++v) {
        float val = acc[m][t][v] + bv;
        if (relu) val = fmaxf(val, 0.0f);
        if constexpr (OUT_BF16)
          ((ushort*)Cout)[(size_t)(mBase + v) * ldc + ncol] = f2bf(val);
        else
          ((float*)Cout)[(size_t)(mBase + v) * ldc + ncol] = val;
      }
    }
  }
}

// ---------------------------------------------------------------------------
// Degree / normalization
// ---------------------------------------------------------------------------
__global__ void k_init_deg(float* __restrict__ deg, int n) {
  int i = blockIdx.x * blockDim.x + threadIdx.x;
  if (i < n) deg[i] = 1.0f;                      // self-loop
}

__global__ void k_count_deg(const int* __restrict__ col,
                            float* __restrict__ deg, int e) {
  int i = blockIdx.x * blockDim.x + threadIdx.x;
  if (i < e) atomicAdd(&deg[col[i]], 1.0f);
}

__global__ void k_rsqrt_inplace(float* __restrict__ deg, int n) {
  int i = blockIdx.x * blockDim.x + threadIdx.x;
  if (i < n) deg[i] = rsqrtf(deg[i]);            // deg buffer becomes dis
}

// acc[n,d] = xw[n,d] * dis[n]^2   (self-loop term; also initializes acc)
__global__ void k_selfloop(const float* __restrict__ xw,
                           const float* __restrict__ dis,
                           float* __restrict__ acc) {
  unsigned i = blockIdx.x * blockDim.x + threadIdx.x;
  if (i >= (unsigned)GN_N * GN_D) return;
  unsigned n = i >> 6;
  float w = dis[n];
  acc[i] = xw[i] * (w * w);
}

// one thread per (edge, channel):  acc[col] += xw[row] * dis[row]*dis[col]
__global__ void k_edge_scatter(const int* __restrict__ row,
                               const int* __restrict__ col,
                               const float* __restrict__ dis,
                               const float* __restrict__ xw,
                               float* __restrict__ acc) {
  unsigned gid = blockIdx.x * blockDim.x + threadIdx.x;
  if (gid >= (unsigned)GN_E * GN_D) return;
  unsigned e = gid >> 6;
  unsigned d = gid & 63u;
  if (d == 0 && e + 2048 < (unsigned)GN_E) {
    __builtin_prefetch(&row[e + 2048], 0, 0);   // global_prefetch_b8
    __builtin_prefetch(&col[e + 2048], 0, 0);
  }
  int r = row[e];
  int c = col[e];
  float w = dis[r] * dis[c];
  atomicAdd(&acc[(size_t)c * GN_D + d], xw[(size_t)r * GN_D + d] * w);
}

// BN(eval) + ReLU + optional residual; writes bf16 into concatenated
// localbf[N, L*D] (GEMM input for next layer and for the encoder)
__global__ void k_bn_relu_skip(const float* __restrict__ acc,
                               const float* __restrict__ cb,
                               const float* __restrict__ gamma,
                               const float* __restrict__ beta,
                               const float* __restrict__ mean,
                               const float* __restrict__ var,
                               ushort* __restrict__ localbf,
                               int layer, int do_skip) {
  unsigned i = blockIdx.x * blockDim.x + threadIdx.x;
  if (i >= (unsigned)GN_N * GN_D) return;
  unsigned n = i >> 6;
  unsigned d = i & 63u;
  float scale = gamma[d] * rsqrtf(var[d] + GN_EPS);
  float v = (acc[i] + cb[d] - mean[d]) * scale + beta[d];
  v = fmaxf(v, 0.0f);
  size_t base = (size_t)n * (GN_L * GN_D);
  if (do_skip) v += bf2f(localbf[base + (size_t)(layer - 2) * GN_D + d]);
  localbf[base + (size_t)layer * GN_D + d] = f2bf(v);
}

// ---------------------------------------------------------------------------
// Graph pooling (scatter-mean over batch)
// ---------------------------------------------------------------------------
__global__ void k_zero(float* __restrict__ p, int n) {
  int i = blockIdx.x * blockDim.x + threadIdx.x;
  if (i < n) p[i] = 0.0f;
}

__global__ void k_pool_sum(const float* __restrict__ h2,
                           const int* __restrict__ batch,
                           float* __restrict__ gsum,
                           float* __restrict__ cnt) {
  unsigned i = blockIdx.x * blockDim.x + threadIdx.x;
  if (i >= (unsigned)GN_N * GN_H2) return;
  unsigned n = i >> 7;
  unsigned d = i & 127u;
  int g = batch[n];
  atomicAdd(&gsum[(size_t)g * GN_H2 + d], h2[i]);
  if (d == 0) atomicAdd(&cnt[g], 1.0f);
}

// mean, written as bf16 (GEMM input for decoder layer 1)
__global__ void k_pool_mean(const float* __restrict__ gsum,
                            const float* __restrict__ cnt,
                            ushort* __restrict__ gfeatbf) {
  int i = blockIdx.x * blockDim.x + threadIdx.x;
  if (i >= GN_G * GN_H2) return;
  int g = i >> 7;
  gfeatbf[i] = f2bf(gsum[i] / fmaxf(cnt[g], 1.0f));
}

// final:  out[g] = dh[g,:] . dec_W2 + dec_b2
__global__ void k_dec2(const float* __restrict__ dh,
                       const float* __restrict__ W2,
                       const float* __restrict__ b2,
                       float* __restrict__ out) {
  int g = blockIdx.x * blockDim.x + threadIdx.x;
  if (g >= GN_G) return;
  float s = b2[0];
#pragma unroll
  for (int d = 0; d < GN_D; ++d) s += dh[(size_t)g * GN_D + d] * W2[d];
  out[g] = s;
}

// ---------------------------------------------------------------------------
extern "C" void kernel_launch(void* const* d_in, const int* in_sizes, int n_in,
                              void* d_out, int out_size, void* d_ws, size_t ws_size,
                              hipStream_t stream) {
  (void)in_sizes; (void)n_in; (void)out_size; (void)ws_size;

  const float* x     = (const float*)d_in[0];
  const int*   ei    = (const int*)  d_in[1];
  const int*   batch = (const int*)  d_in[2];
  const float* convW = (const float*)d_in[3];
  const float* convb = (const float*)d_in[4];
  const float* gam   = (const float*)d_in[5];
  const float* bet   = (const float*)d_in[6];
  const float* mu    = (const float*)d_in[7];
  const float* var   = (const float*)d_in[8];
  const float* eW1   = (const float*)d_in[9];
  const float* eb1   = (const float*)d_in[10];
  const float* eW2   = (const float*)d_in[11];
  const float* eb2   = (const float*)d_in[12];
  const float* dW1   = (const float*)d_in[13];
  const float* db1   = (const float*)d_in[14];
  const float* dW2   = (const float*)d_in[15];
  const float* db2   = (const float*)d_in[16];

  const int* rowp = ei;          // source
  const int* colp = ei + GN_E;   // target

  // Workspace carve-up
  char*  ws  = (char*)d_ws;
  size_t off = 0;
  auto carve = [&](size_t bytes) -> void* {
    void* p = ws + off;
    off = (off + bytes + 255) & ~(size_t)255;
    return p;
  };
  float*  dis     = (float*)carve((size_t)GN_N * 4);
  float*  xw      = (float*)carve((size_t)GN_N * GN_D * 4);
  float*  acc     = (float*)carve((size_t)GN_N * GN_D * 4);
  float*  h2      = (float*)carve((size_t)GN_N * GN_H2 * 4);
  float*  gsum    = (float*)carve((size_t)GN_G * GN_H2 * 4);
  float*  cnt     = (float*)carve((size_t)GN_G * 4);
  float*  dh      = (float*)carve((size_t)GN_G * GN_D * 4);
  ushort* xbf     = (ushort*)carve((size_t)GN_N * GN_D * 2);
  ushort* localbf = (ushort*)carve((size_t)GN_N * GN_L * GN_D * 2);
  ushort* h1bf    = (ushort*)carve((size_t)GN_N * GN_H1 * 2);
  ushort* gfeatbf = (ushort*)carve((size_t)GN_G * GN_H2 * 2);
  ushort* pConvW  = (ushort*)carve((size_t)GN_L * GN_D * GN_D * 2);
  ushort* pEW1    = (ushort*)carve((size_t)GN_L * GN_D * GN_H1 * 2);
  ushort* pEW2    = (ushort*)carve((size_t)GN_H1 * GN_H2 * 2);
  ushort* pDW1    = (ushort*)carve((size_t)GN_H2 * GN_D * 2);

  const int T = 256;

  // ---- one-time conversions / weight packing (cheap, deterministic) ------
  k_f2bf<<<(GN_N * GN_D + T - 1) / T, T, 0, stream>>>(x, xbf, GN_N * GN_D);
  for (int l = 0; l < GN_L; ++l)
    k_pack_w<<<(GN_D * GN_D + T - 1) / T, T, 0, stream>>>(
        convW + (size_t)l * GN_D * GN_D, pConvW + (size_t)l * GN_D * GN_D,
        GN_D, GN_D);
  k_pack_w<<<(GN_L * GN_D * GN_H1 + T - 1) / T, T, 0, stream>>>(
      eW1, pEW1, GN_L * GN_D, GN_H1);
  k_pack_w<<<(GN_H1 * GN_H2 + T - 1) / T, T, 0, stream>>>(
      eW2, pEW2, GN_H1, GN_H2);
  k_pack_w<<<(GN_H2 * GN_D + T - 1) / T, T, 0, stream>>>(
      dW1, pDW1, GN_H2, GN_D);

  // ---- GCN normalization -------------------------------------------------
  k_init_deg<<<(GN_N + T - 1) / T, T, 0, stream>>>(dis, GN_N);
  k_count_deg<<<(GN_E + T - 1) / T, T, 0, stream>>>(colp, dis, GN_E);
  k_rsqrt_inplace<<<(GN_N + T - 1) / T, T, 0, stream>>>(dis, GN_N);

  // ---- conv layers -------------------------------------------------------
  const unsigned nd = (unsigned)GN_N * GN_D;
  const unsigned ed = (unsigned)GN_E * GN_D;
  for (int l = 0; l < GN_L; ++l) {
    const ushort* Ain = (l == 0) ? xbf : (localbf + (size_t)(l - 1) * GN_D);
    const int     lda = (l == 0) ? GN_D : GN_L * GN_D;
    wmma_gemm_bf16<false><<<dim3(GN_N / 32, GN_D / 64), 32, 0, stream>>>(
        Ain, lda, pConvW + (size_t)l * GN_D * GN_D, nullptr,
        xw, GN_D, GN_D, /*relu=*/0);
    k_selfloop<<<(nd + T - 1) / T, T, 0, stream>>>(xw, dis, acc);
    k_edge_scatter<<<(ed + T - 1) / T, T, 0, stream>>>(rowp, colp, dis, xw, acc);
    k_bn_relu_skip<<<(nd + T - 1) / T, T, 0, stream>>>(
        acc, convb + l * GN_D, gam + l * GN_D, bet + l * GN_D,
        mu + l * GN_D, var + l * GN_D, localbf, l, (l == 2) ? 1 : 0);
  }

  // ---- encoder -----------------------------------------------------------
  wmma_gemm_bf16<true><<<dim3(GN_N / 32, GN_H1 / 64), 32, 0, stream>>>(
      localbf, GN_L * GN_D, pEW1, eb1, h1bf, GN_H1, GN_L * GN_D, /*relu=*/1);
  wmma_gemm_bf16<false><<<dim3(GN_N / 32, GN_H2 / 64), 32, 0, stream>>>(
      h1bf, GN_H1, pEW2, eb2, h2, GN_H2, GN_H1, /*relu=*/1);

  // ---- graph pooling (scatter mean) --------------------------------------
  k_zero<<<(GN_G * GN_H2 + T - 1) / T, T, 0, stream>>>(gsum, GN_G * GN_H2);
  k_zero<<<1, GN_G, 0, stream>>>(cnt, GN_G);
  const unsigned nh2 = (unsigned)GN_N * GN_H2;
  k_pool_sum<<<(nh2 + T - 1) / T, T, 0, stream>>>(h2, batch, gsum, cnt);
  k_pool_mean<<<(GN_G * GN_H2 + T - 1) / T, T, 0, stream>>>(gsum, cnt, gfeatbf);

  // ---- decoder -----------------------------------------------------------
  wmma_gemm_bf16<false><<<dim3(GN_G / 32, GN_D / 64), 32, 0, stream>>>(
      gfeatbf, GN_H2, pDW1, db1, dh, GN_D, GN_H2, /*relu=*/1);
  k_dec2<<<1, GN_G, 0, stream>>>(dh, dW2, db2, (float*)d_out);
}
